// LigandGNN_67929202754018
// MI455X (gfx1250) — matrix operand order repaired and checked by reading, not compiled, not run on hardware
//
#include <hip/hip_runtime.h>
#include <hip/hip_bf16.h>

// ---------------------------------------------------------------------------
// LigandGNN on gfx1250: GraphConv x4 (+BN+ReLU) -> mean pool -> MLP head
//   - edge aggregation: wave32-per-edge, global_atomic_add_f32 (L2-resident)
//   - node GEMMs:      v_wmma_f32_16x16x32_f16 (f16 in, f32 accumulate),
//                      2 N-tiles per wave to amortize A-fragment load+cvt
// ---------------------------------------------------------------------------

#define GN 50000
#define GE 500000
#define GF 128
#define GH 192
#define GG 256
#define MTILES (GN / 16)   // 3125
#define NTILES (GH / 16)   // 12
#define NPAIRS (NTILES / 2) // 6: each wave computes 2 adjacent N-tiles

typedef __attribute__((ext_vector_type(16))) _Float16 v16h;
typedef __attribute__((ext_vector_type(8)))  float    v8f;

// ---------------------- edge scatter-add (segment_sum) ----------------------
__global__ void scatter_add_kernel(const float* __restrict__ x,
                                   const int* __restrict__ src,
                                   const int* __restrict__ dst,
                                   float* __restrict__ agg, int D) {
    int wave = (blockIdx.x * blockDim.x + threadIdx.x) >> 5;
    int lane = threadIdx.x & 31;
    if (wave >= GE) return;
    int s = src[wave];
    int d = dst[wave];
    const float* xs = x + (size_t)s * D;
    float*       ad = agg + (size_t)d * D;
    for (int c = lane; c < D; c += 32) {
        atomicAdd(ad + c, xs[c]);     // global_atomic_add_f32, no return
    }
}

// ------------------------- WMMA fragment loaders ----------------------------
// A: 16x32 f16 tile from row-major f32 [rows x ld].
// Lane layout (ISA 7.12.2): lanes 0-15 M=lane, K pairs {0..7,16..23};
//                           lanes 16-31 M=lane-16, K pairs {8..15,24..31}.
__device__ __forceinline__ v16h load_frag_a(const float* __restrict__ base,
                                            int ld, int row, int k0, int lane) {
    int kb = (lane >> 4) << 3;                 // 0 or 8
    const float* p = base + (size_t)row * ld + k0 + kb;
    v16h out;
#pragma unroll
    for (int i = 0; i < 8; ++i) {
        int kk = (i < 4) ? (2 * i) : (8 + 2 * i);   // 0,2,4,6,16,18,20,22
        float2 v = *(const float2*)(p + kk);
        out[2 * i]     = (_Float16)v.x;
        out[2 * i + 1] = (_Float16)v.y;
    }
    return out;
}

// B: 32x16 f16 tile where B[k][n] = W[n][k], W row-major f32 [GH x ld].
// Lane = N; lanes 0-15 hold K {0..7,16..23}, lanes 16-31 hold K {8..15,24..31}.
__device__ __forceinline__ v16h load_frag_b(const float* __restrict__ w,
                                            int ld, int ncol, int k0, int lane) {
    int kb = (lane >> 4) << 3;                 // 0 or 8
    const float* p = w + (size_t)ncol * ld + k0 + kb;
    v16h out;
#pragma unroll
    for (int i = 0; i < 8; ++i) {
        float2 v = *(const float2*)(p + ((i < 4) ? (2 * i) : (8 + 2 * i)));
        out[2 * i]     = (_Float16)v.x;
        out[2 * i + 1] = (_Float16)v.y;
    }
    return out;
}

// ------------- fused dual GEMM: H = Aagg*Wrel^T + Aroot*Wroot^T + b ---------
// Each wave owns one M-tile and TWO adjacent N-tiles: A fragments are loaded
// and converted once and feed two v_wmma each.
__global__ void gemm_dual_kernel(const float* __restrict__ Aagg,
                                 const float* __restrict__ Aroot,
                                 const float* __restrict__ Wrel,
                                 const float* __restrict__ Wroot,
                                 const float* __restrict__ bias,
                                 float* __restrict__ Hout, int K) {
    int wid = (blockIdx.x * blockDim.x + threadIdx.x) >> 5;
    if (wid >= MTILES * NPAIRS) return;        // wave-uniform guard: EXEC stays full
    int lane  = threadIdx.x & 31;
    int mtile = wid / NPAIRS;
    int npair = wid % NPAIRS;
    int arow  = mtile * 16 + (lane & 15);
    int ncol0 = npair * 32 + (lane & 15);      // N-tile 2*npair
    int ncol1 = ncol0 + 16;                    // N-tile 2*npair+1

    v8f acc0 = {};
    v8f acc1 = {};
    for (int k0 = 0; k0 < K; k0 += 32) {
        // aggregated-neighbor term
        v16h a0  = load_frag_a(Aagg, K, arow, k0, lane);
        v16h b00 = load_frag_b(Wrel, K, ncol0, k0, lane);
        v16h b01 = load_frag_b(Wrel, K, ncol1, k0, lane);
        acc0 = __builtin_amdgcn_wmma_f32_16x16x32_f16(false, a0, false, b00,
                                                      (short)0, acc0, false, false);
        acc1 = __builtin_amdgcn_wmma_f32_16x16x32_f16(false, a0, false, b01,
                                                      (short)0, acc1, false, false);
        // root (self) term
        v16h a1  = load_frag_a(Aroot, K, arow, k0, lane);
        v16h b10 = load_frag_b(Wroot, K, ncol0, k0, lane);
        v16h b11 = load_frag_b(Wroot, K, ncol1, k0, lane);
        acc0 = __builtin_amdgcn_wmma_f32_16x16x32_f16(false, a1, false, b10,
                                                      (short)0, acc0, false, false);
        acc1 = __builtin_amdgcn_wmma_f32_16x16x32_f16(false, a1, false, b11,
                                                      (short)0, acc1, false, false);
    }

    float bv0   = bias[ncol0];
    float bv1   = bias[ncol1];
    int   rbase = mtile * 16 + ((lane >> 4) << 3);   // M = r + (lane>=16 ? 8:0)
#pragma unroll
    for (int r = 0; r < 8; ++r) {
        float* rowp = Hout + (size_t)(rbase + r) * GH;
        rowp[ncol0] = acc0[r] + bv0;
        rowp[ncol1] = acc1[r] + bv1;
    }
}

// ------------------------- BN statistics (per channel) ----------------------
__global__ void bn_stats_kernel(const float* __restrict__ h,
                                float* __restrict__ sums,
                                float* __restrict__ sqs) {
    int c = threadIdx.x;                       // 0..GH-1
    float s = 0.f, q = 0.f;
    for (int r = blockIdx.x; r < GN; r += gridDim.x) {
        float v = h[(size_t)r * GH + c];
        s += v;
        q += v * v;
    }
    atomicAdd(&sums[c], s);
    atomicAdd(&sqs[c], q);
}

// ------------------------- BN apply + ReLU (in place) -----------------------
__global__ void bn_apply_kernel(float* __restrict__ h,
                                const float* __restrict__ sums,
                                const float* __restrict__ sqs,
                                const float* __restrict__ gamma,
                                const float* __restrict__ beta) {
    size_t i = (size_t)blockIdx.x * blockDim.x + threadIdx.x;
    if (i >= (size_t)GN * GH) return;
    int   c   = (int)(i % GH);
    float mu  = sums[c] * (1.0f / GN);
    float var = sqs[c] * (1.0f / GN) - mu * mu;
    float sc  = rsqrtf(var + 1e-5f) * gamma[c];
    float v   = (h[i] - mu) * sc + beta[c];
    h[i] = v > 0.f ? v : 0.f;
}

// ----------------------------- mean pool ------------------------------------
__global__ void pool_sum_kernel(const float* __restrict__ h,
                                const int* __restrict__ batch,
                                float* __restrict__ pooled,
                                float* __restrict__ cnt) {
    int node = (blockIdx.x * blockDim.x + threadIdx.x) >> 5;
    int lane = threadIdx.x & 31;
    if (node >= GN) return;
    int b = batch[node];
    const float* hp = h + (size_t)node * GH;
    float*       pp = pooled + (size_t)b * GH;
    for (int c = lane; c < GH; c += 32) atomicAdd(pp + c, hp[c]);
    if (lane == 0) atomicAdd(&cnt[b], 1.0f);
}

// ----------------------------- MLP head -------------------------------------
__global__ void head_kernel(const float* __restrict__ pooled,
                            const float* __restrict__ cnt,
                            const float* __restrict__ w1, const float* __restrict__ b1,
                            const float* __restrict__ w2, const float* __restrict__ b2,
                            const float* __restrict__ ow, const float* __restrict__ ob,
                            float* __restrict__ out) {
    __shared__ float g[GH], h1[GH], red[GH];
    int t    = threadIdx.x;                    // 0..GH-1
    int gidx = blockIdx.x;
    float c = cnt[gidx];
    c = c < 1.f ? 1.f : c;
    g[t] = pooled[(size_t)gidx * GH + t] / c;
    __syncthreads();
    float a = b1[t];
    for (int k = 0; k < GH; ++k) a += g[k] * w1[t * GH + k];
    h1[t] = a > 0.f ? a : 0.f;
    __syncthreads();
    float b = b2[t];
    for (int k = 0; k < GH; ++k) b += h1[k] * w2[t * GH + k];
    red[t] = b * ow[t];
    __syncthreads();
    if (t == 0) {
        float tot = ob[0];
        for (int k = 0; k < GH; ++k) tot += red[k];
        out[gidx] = tot;
    }
}

// ---------------------------------------------------------------------------
extern "C" void kernel_launch(void* const* d_in, const int* in_sizes, int n_in,
                              void* d_out, int out_size, void* d_ws, size_t ws_size,
                              hipStream_t stream) {
    (void)in_sizes; (void)n_in; (void)out_size; (void)ws_size;

    const float* x       = (const float*)d_in[0];
    const int*   ei      = (const int*)d_in[1];
    const int*   src     = ei;
    const int*   dst     = ei + GE;
    const int*   batch   = (const int*)d_in[2];
    // d_in[3] = num_graphs (compile-time constant GG)
    const float* w_rel0  = (const float*)d_in[4];
    const float* b_rel0  = (const float*)d_in[5];
    const float* w_root0 = (const float*)d_in[6];
    const float* w_rel   = (const float*)d_in[7];
    const float* b_rel   = (const float*)d_in[8];
    const float* w_root  = (const float*)d_in[9];
    const float* gamma   = (const float*)d_in[10];
    const float* beta    = (const float*)d_in[11];
    const float* head_w1 = (const float*)d_in[12];
    const float* head_b1 = (const float*)d_in[13];
    const float* head_w2 = (const float*)d_in[14];
    const float* head_b2 = (const float*)d_in[15];
    const float* out_w   = (const float*)d_in[16];
    const float* out_b   = (const float*)d_in[17];
    float*       out     = (float*)d_out;

    // workspace layout (floats)
    float* ws     = (float*)d_ws;
    float* actA   = ws;                                  // N*H
    float* actB   = actA + (size_t)GN * GH;              // N*H
    float* agg    = actB + (size_t)GN * GH;              // N*H (K<=H)
    float* sums   = agg + (size_t)GN * GH;               // H
    float* sqs    = sums + GH;                           // H
    float* pooled = sqs + GH;                            // G*H
    float* cnt    = pooled + (size_t)GG * GH;            // G

    const int scatterBlocks = (GE * 32) / 256;                   // 62500
    const int gemmBlocks    = (MTILES * NPAIRS + 7) / 8;         // 2344 (8 waves/blk)
    const int bnBlocks      = (GN * GH + 255) / 256;
    const int poolBlocks    = (GN * 32 + 255) / 256;

    auto layer = [&](const float* in, float* outAct, int K,
                     const float* wr, const float* br, const float* wrt,
                     const float* gm, const float* bt) {
        hipMemsetAsync(agg, 0, (size_t)GN * K * sizeof(float), stream);
        hipMemsetAsync(sums, 0, 2 * GH * sizeof(float), stream); // sums+sqs contiguous
        scatter_add_kernel<<<scatterBlocks, 256, 0, stream>>>(in, src, dst, agg, K);
        gemm_dual_kernel<<<gemmBlocks, 256, 0, stream>>>(agg, in, wr, wrt, br, outAct, K);
        bn_stats_kernel<<<512, GH, 0, stream>>>(outAct, sums, sqs);
        bn_apply_kernel<<<bnBlocks, 256, 0, stream>>>(outAct, sums, sqs, gm, bt);
    };

    layer(x,    actA, GF, w_rel0,              b_rel0,        w_root0,
          gamma + 0 * GH, beta + 0 * GH);
    layer(actA, actB, GH, w_rel + 0 * GH * GH, b_rel + 0 * GH, w_root + 0 * GH * GH,
          gamma + 1 * GH, beta + 1 * GH);
    layer(actB, actA, GH, w_rel + 1 * GH * GH, b_rel + 1 * GH, w_root + 1 * GH * GH,
          gamma + 2 * GH, beta + 2 * GH);
    layer(actA, actB, GH, w_rel + 2 * GH * GH, b_rel + 2 * GH, w_root + 2 * GH * GH,
          gamma + 3 * GH, beta + 3 * GH);

    hipMemsetAsync(pooled, 0, ((size_t)GG * GH + GG) * sizeof(float), stream);
    pool_sum_kernel<<<poolBlocks, 256, 0, stream>>>(actB, batch, pooled, cnt);
    head_kernel<<<GG, GH, 0, stream>>>(pooled, cnt, head_w1, head_b1,
                                       head_w2, head_b2, out_w, out_b, out);
}
